// PointNetSetAbstraction_21457656611508
// MI455X (gfx1250) — matrix operand order repaired and checked by reading, not compiled, not run on hardware
//
#include <hip/hip_runtime.h>
#include <hip/hip_bf16.h>

typedef _Float16 v16h __attribute__((ext_vector_type(16)));
typedef _Float16 v8h  __attribute__((ext_vector_type(8)));
typedef float    v8f  __attribute__((ext_vector_type(8)));

#define NPTS   4096
#define NBATCH 16
#define SPTS   512
#define KSAMP  32
#define MROWS  (NBATCH * SPTS * KSAMP)   // 262144
#define MTILES (MROWS / 16)              // 16384
#define R2     0.04f

// ---------------------------------------------------------------------------
// 1) Farthest point sampling: one workgroup per batch, dist array in LDS.
//    Emits centroid s = `far` BEFORE the update (matches the reference scan),
//    argmax prefers the lowest index on ties (matches jnp.argmax).
// ---------------------------------------------------------------------------
__global__ __launch_bounds__(256)
void fps_kernel(const float* __restrict__ xyz, float* __restrict__ newxyz_ws,
                float* __restrict__ out_xyz)
{
    __shared__ float ldist[NPTS];
    __shared__ float rval[256];
    __shared__ int   ridx[256];
    __shared__ int   sfar;
    const int b   = blockIdx.x;
    const int tid = threadIdx.x;
    const float* xb = xyz + (size_t)b * 3 * NPTS;

    for (int n = tid; n < NPTS; n += 256) ldist[n] = 1e10f;
    __syncthreads();

    int far = 0;
    for (int s = 0; s < SPTS; s++) {
        const float cx = xb[far], cy = xb[NPTS + far], cz = xb[2 * NPTS + far];
        if (tid == 0) {
            float* nx = newxyz_ws + ((size_t)b * SPTS + s) * 3;
            nx[0] = cx; nx[1] = cy; nx[2] = cz;
            float* ox = out_xyz + (size_t)b * 3 * SPTS;
            ox[s] = cx; ox[SPTS + s] = cy; ox[2 * SPTS + s] = cz;
        }
        float bv = -1.0f; int bi = 0;
        for (int n = tid; n < NPTS; n += 256) {
            const float dx = xb[n] - cx, dy = xb[NPTS + n] - cy, dz = xb[2 * NPTS + n] - cz;
            const float d  = dx * dx + dy * dy + dz * dz;
            const float dd = fminf(ldist[n], d);
            ldist[n] = dd;
            if (dd > bv) { bv = dd; bi = n; }
        }
        rval[tid] = bv; ridx[tid] = bi;
        __syncthreads();
        for (int off = 128; off > 0; off >>= 1) {
            if (tid < off) {
                const float v2 = rval[tid + off]; const int i2 = ridx[tid + off];
                if (v2 > rval[tid] || (v2 == rval[tid] && i2 < ridx[tid])) {
                    rval[tid] = v2; ridx[tid] = i2;
                }
            }
            __syncthreads();
        }
        if (tid == 0) sfar = ridx[0];
        __syncthreads();
        far = sfar;
        __syncthreads();
    }
}

// ---------------------------------------------------------------------------
// 2) Weight repack: f32 [Cout,Cin] -> f16 [Cout,Kt] zero-padded (WMMA B rows).
// ---------------------------------------------------------------------------
__global__ void wprep_kernel(const float* __restrict__ W, _Float16* __restrict__ Wh,
                             int Cout, int Cin, int Kt)
{
    const int t = blockIdx.x * blockDim.x + threadIdx.x;
    if (t >= Cout * Kt) return;
    const int c = t / Kt, k = t % Kt;
    Wh[t] = (k < Cin) ? (_Float16)W[c * Cin + k] : (_Float16)0.0f;
}

// ---------------------------------------------------------------------------
// 3) Ball query + grouping: one thread per (b,s); ascending scan == sorted
//    first-K semantics of the reference; pad misses with the first hit.
//    Output: f16 h0 rows [M, 32] (cols 0..5 data, 6..31 zero K-padding).
// ---------------------------------------------------------------------------
__global__ __launch_bounds__(256)
void group_kernel(const float* __restrict__ xyz, const float* __restrict__ pts,
                  const float* __restrict__ newxyz, _Float16* __restrict__ h0)
{
    const int t = blockIdx.x * blockDim.x + threadIdx.x;
    if (t >= NBATCH * SPTS) return;
    const int b = t >> 9;
    const float* xb = xyz + (size_t)b * 3 * NPTS;
    const float* pb = pts + (size_t)b * 3 * NPTS;
    const float cx = newxyz[t * 3 + 0], cy = newxyz[t * 3 + 1], cz = newxyz[t * 3 + 2];

    int gi[KSAMP];
    int cnt = 0;
    for (int n = 0; n < NPTS && cnt < KSAMP; n++) {
        const float dx = xb[n] - cx, dy = xb[NPTS + n] - cy, dz = xb[2 * NPTS + n] - cz;
        if (dx * dx + dy * dy + dz * dz <= R2) gi[cnt++] = n;
    }
    if (cnt == 0) { gi[0] = 0; cnt = 1; }   // centroid itself always qualifies
    for (int k = cnt; k < KSAMP; k++) gi[k] = gi[0];

    _Float16* row = h0 + (size_t)t * KSAMP * 32;
    for (int k = 0; k < KSAMP; k++) {
        const int n = gi[k];
        _Float16* rr = row + k * 32;
        rr[0] = (_Float16)(xb[n] - cx);
        rr[1] = (_Float16)(xb[NPTS + n] - cy);
        rr[2] = (_Float16)(xb[2 * NPTS + n] - cz);
        rr[3] = (_Float16)pb[n];
        rr[4] = (_Float16)pb[NPTS + n];
        rr[5] = (_Float16)pb[2 * NPTS + n];
        #pragma unroll
        for (int e = 6; e < 32; e++) rr[e] = (_Float16)0.0f;
    }
}

// ---------------------------------------------------------------------------
// 4) WMMA GEMM + fused input BN/ReLU + BN-stat accumulation.
//    One wave owns MPW consecutive 16-row M-tiles of one 16-col N-tile:
//      - B (weight) fragments loaded once, reused across all MPW tiles
//      - BN affine coefficients hoisted per K-step, reused across tiles
//      - MPW independent accumulators -> back-to-back WMMA issue (no
//        WMMA->WMMA RAW hazard NOPs, per ISA 7.12.1)
//      - USE_AFF is a compile-time parameter: no branch inside WMMA region
//    A frag (16-bit, 16x32): lane<16 -> K {kk..kk+7, kk+16..kk+23} of row m0+lane,
//                            lane>=16 -> K {kk+8..kk+15, kk+24..kk+31}.
//    B frag: same K order as A, from weight row (n0+lane).
// ---------------------------------------------------------------------------
template<int KSTEPS, int NTILES, int MPW, bool USE_AFF>
__global__ __launch_bounds__(256)
void gemm_bn_kernel(const _Float16* __restrict__ A, int lda,
                    const _Float16* __restrict__ Wh,
                    const float* __restrict__ bias,
                    const float* __restrict__ affa, const float* __restrict__ affb,
                    _Float16* __restrict__ Z, int Cout,
                    float* __restrict__ sum, float* __restrict__ sumsq)
{
    constexpr int MGROUPS = MTILES / MPW;
    const int wid = blockIdx.x * 8 + (threadIdx.x >> 5);
    if (wid >= MGROUPS * NTILES) return;          // wave-uniform: EXEC stays all-1s
    const int lane = threadIdx.x & 31;
    const int nt = wid % NTILES;
    const int mg = wid / NTILES;
    const int n0 = nt * 16;
    const int l16 = lane & 15;
    const int hi  = (lane >> 4) & 1;
    const int col = n0 + l16;
    const int ko  = hi * 8;
    constexpr int ldw = KSTEPS * 32;

    // speculative prefetch of the next M-group's activations (OOB-safe)
    __builtin_prefetch(A + ((size_t)((mg + 1) * MPW * 16 + l16)) * lda, 0, 1);

    // weight fragments: loaded once per wave, reused across MPW m-tiles
    v16h bf[KSTEPS];
    #pragma unroll
    for (int ks = 0; ks < KSTEPS; ks++) {
        const v8h blo = *(const v8h*)(Wh + (size_t)col * ldw + ks * 32 + hi * 8);
        const v8h bhi = *(const v8h*)(Wh + (size_t)col * ldw + ks * 32 + 16 + hi * 8);
        bf[ks] = __builtin_shufflevector(blo, bhi, 0, 1, 2, 3, 4, 5, 6, 7, 8, 9, 10, 11, 12, 13, 14, 15);
    }

    const float cb = bias[col];
    v8f acc[MPW];
    #pragma unroll
    for (int mi = 0; mi < MPW; mi++)
        #pragma unroll
        for (int j = 0; j < 8; j++) acc[mi][j] = cb;

    #pragma unroll
    for (int ks = 0; ks < KSTEPS; ks++) {
        const int kk = ks * 32;

        // hoisted BN affine coefficients: depend only on K-column, shared by
        // all MPW tiles in this wave
        float aa0[8], ab0[8], aa1[8], ab1[8];
        if constexpr (USE_AFF) {
            #pragma unroll
            for (int e = 0; e < 8; e++) {
                aa0[e] = affa[kk + ko + e];       ab0[e] = affb[kk + ko + e];
                aa1[e] = affa[kk + 16 + ko + e];  ab1[e] = affb[kk + 16 + ko + e];
            }
        }

        #pragma unroll
        for (int mi = 0; mi < MPW; mi++) {
            const int r = (mg * MPW + mi) * 16 + l16;
            const _Float16* ap = A + (size_t)r * lda + kk + ko;
            const v8h alo = *(const v8h*)ap;
            const v8h ahi = *(const v8h*)(ap + 16);
            v16h af;
            if constexpr (USE_AFF) {
                #pragma unroll
                for (int e = 0; e < 8; e++) {
                    af[e]     = (_Float16)fmaxf(aa0[e] * (float)alo[e] + ab0[e], 0.0f);
                    af[8 + e] = (_Float16)fmaxf(aa1[e] * (float)ahi[e] + ab1[e], 0.0f);
                }
            } else {
                #pragma unroll
                for (int e = 0; e < 8; e++) { af[e] = alo[e]; af[8 + e] = ahi[e]; }
            }
            acc[mi] = __builtin_amdgcn_wmma_f32_16x16x32_f16(false, af, false, bf[ks],
                                                             (short)0, acc[mi], false, false);
        }
    }

    // per-channel sum / sumsq for training-mode BN of this layer's output,
    // combined across the MPW tiles (same column set) before the atomics.
    float s = 0.0f, q = 0.0f;
    #pragma unroll
    for (int mi = 0; mi < MPW; mi++)
        #pragma unroll
        for (int j = 0; j < 8; j++) { s += acc[mi][j]; q += acc[mi][j] * acc[mi][j]; }
    s += __shfl_xor(s, 16, 32);
    q += __shfl_xor(q, 16, 32);
    if (lane < 16) { atomicAdd(&sum[col], s); atomicAdd(&sumsq[col], q); }

    // store raw z (pre-BN) as f16
    #pragma unroll
    for (int mi = 0; mi < MPW; mi++) {
        const int m0 = (mg * MPW + mi) * 16;
        #pragma unroll
        for (int j = 0; j < 8; j++) {
            const int row = m0 + hi * 8 + j;
            Z[(size_t)row * Cout + col] = (_Float16)acc[mi][j];
        }
    }
}

// ---------------------------------------------------------------------------
// 5) BN stats -> per-channel affine (a = g*rsqrt(var+eps), b = be - mean*a)
// ---------------------------------------------------------------------------
__global__ void bn_stats_kernel(const float* __restrict__ sum, const float* __restrict__ sumsq,
                                const float* __restrict__ g, const float* __restrict__ be,
                                float* __restrict__ affa, float* __restrict__ affb,
                                int C, float invM)
{
    const int c = blockIdx.x * blockDim.x + threadIdx.x;
    if (c >= C) return;
    const float m = sum[c] * invM;
    const float v = fmaxf(sumsq[c] * invM - m * m, 0.0f);
    const float a = g[c] * rsqrtf(v + 1e-5f);
    affa[c] = a;
    affb[c] = be[c] - m * a;
}

// ---------------------------------------------------------------------------
// 6) Final: BN+ReLU fold on z3, max over K=32, write [B,128,S] transposed.
// ---------------------------------------------------------------------------
__global__ __launch_bounds__(256)
void final_max_kernel(const _Float16* __restrict__ z3,
                      const float* __restrict__ affa, const float* __restrict__ affb,
                      float* __restrict__ out_points)
{
    const int t = blockIdx.x * blockDim.x + threadIdx.x;
    if (t >= NBATCH * SPTS * 128) return;
    const int c = t & 127;
    const int s = (t >> 7) & 511;
    const int b = t >> 16;
    const size_t mb = ((size_t)(b * SPTS + s)) * KSAMP;
    const float a = affa[c], bb = affb[c];
    float mx = 0.0f;                       // ReLU floors all values at 0
    for (int k = 0; k < KSAMP; k++) {
        const float z = (float)z3[(mb + k) * 128 + c];
        mx = fmaxf(mx, a * z + bb);
    }
    out_points[(size_t)b * (128 * SPTS) + (size_t)c * SPTS + s] = mx;
}

// ---------------------------------------------------------------------------
extern "C" void kernel_launch(void* const* d_in, const int* in_sizes, int n_in,
                              void* d_out, int out_size, void* d_ws, size_t ws_size,
                              hipStream_t stream)
{
    const float* xyz = (const float*)d_in[0];
    const float* pts = (const float*)d_in[1];
    const float* W0  = (const float*)d_in[2];
    const float* b0  = (const float*)d_in[3];
    const float* g0  = (const float*)d_in[4];
    const float* be0 = (const float*)d_in[5];
    const float* W1  = (const float*)d_in[6];
    const float* b1  = (const float*)d_in[7];
    const float* g1  = (const float*)d_in[8];
    const float* be1 = (const float*)d_in[9];
    const float* W2  = (const float*)d_in[10];
    const float* b2  = (const float*)d_in[11];
    const float* g2  = (const float*)d_in[12];
    const float* be2 = (const float*)d_in[13];
    float* out = (float*)d_out;

    char* ws = (char*)d_ws;
    size_t off = 0;
    auto take = [&](size_t bytes) -> char* {
        char* p = ws + off;
        off = (off + bytes + 255) & ~(size_t)255;
        return p;
    };

    float*    newxyz = (float*)take((size_t)NBATCH * SPTS * 3 * sizeof(float));
    _Float16* h0     = (_Float16*)take((size_t)MROWS * 32  * sizeof(_Float16));
    _Float16* z1     = (_Float16*)take((size_t)MROWS * 64  * sizeof(_Float16));
    _Float16* z2     = (_Float16*)take((size_t)MROWS * 64  * sizeof(_Float16));
    _Float16* z3     = (_Float16*)take((size_t)MROWS * 128 * sizeof(_Float16));
    _Float16* Wh0    = (_Float16*)take((size_t)64  * 32 * sizeof(_Float16));
    _Float16* Wh1    = (_Float16*)take((size_t)64  * 64 * sizeof(_Float16));
    _Float16* Wh2    = (_Float16*)take((size_t)128 * 64 * sizeof(_Float16));
    float*    stats  = (float*)take((size_t)(128 + 128 + 256) * sizeof(float));
    float*    affs   = (float*)take((size_t)(128 + 128 + 256) * sizeof(float));

    const float invM = 1.0f / (float)MROWS;

    hipMemsetAsync(stats, 0, (128 + 128 + 256) * sizeof(float), stream);

    fps_kernel<<<NBATCH, 256, 0, stream>>>(xyz, newxyz, out);
    wprep_kernel<<<(64 * 32 + 255) / 256, 256, 0, stream>>>(W0, Wh0, 64, 6, 32);
    wprep_kernel<<<(64 * 64 + 255) / 256, 256, 0, stream>>>(W1, Wh1, 64, 64, 64);
    wprep_kernel<<<(128 * 64 + 255) / 256, 256, 0, stream>>>(W2, Wh2, 128, 64, 64);
    group_kernel<<<(NBATCH * SPTS) / 256, 256, 0, stream>>>(xyz, pts, newxyz, h0);

    // layer 1: 6(->32 padded) -> 64, no input affine
    gemm_bn_kernel<1, 4, 4, false><<<(MTILES / 4) * 4 / 8, 256, 0, stream>>>(
        h0, 32, Wh0, b0, affs, affs + 64,
        z1, 64, stats + 0, stats + 64);
    bn_stats_kernel<<<1, 128, 0, stream>>>(stats + 0, stats + 64, g0, be0,
                                           affs + 0, affs + 64, 64, invM);

    // layer 2: 64 -> 64, fused BN1+ReLU on input
    gemm_bn_kernel<2, 4, 4, true><<<(MTILES / 4) * 4 / 8, 256, 0, stream>>>(
        z1, 64, Wh1, b1, affs + 0, affs + 64,
        z2, 64, stats + 128, stats + 192);
    bn_stats_kernel<<<1, 128, 0, stream>>>(stats + 128, stats + 192, g1, be1,
                                           affs + 128, affs + 192, 64, invM);

    // layer 3: 64 -> 128, fused BN2+ReLU on input
    gemm_bn_kernel<2, 8, 4, true><<<(MTILES / 4) * 8 / 8, 256, 0, stream>>>(
        z2, 64, Wh2, b2, affs + 128, affs + 192,
        z3, 128, stats + 256, stats + 384);
    bn_stats_kernel<<<1, 128, 0, stream>>>(stats + 256, stats + 384, g2, be2,
                                           affs + 256, affs + 384, 128, invM);

    // BN3+ReLU + max over K, transposed write
    final_max_kernel<<<(NBATCH * SPTS * 128) / 256, 256, 0, stream>>>(
        z3, affs + 256, affs + 384, out + NBATCH * 3 * SPTS);
}
